// GNNDynamicsModel_22703197126861
// MI455X (gfx1250) — matrix operand order repaired and checked by reading, not compiled, not run on hardware
//
#include <hip/hip_runtime.h>
#include <hip/hip_fp16.h>
#include <math.h>

// ---------------- problem constants (match reference) ----------------
#define NNODE 50000
#define NEDGE 600000
#define ETOT  (NNODE + NEDGE)   // with self loops
#define HDIM  128
#define NHEAD 4
#define CDIM  32
#define FEAT_ 32
#define ACT_  8
#define KENC  64                // padded 40 -> 64 for WMMA K multiple of 32
#define NEG_SLOPE_ 0.2f
#define LN_EPS_ 1e-5f

typedef __attribute__((ext_vector_type(16))) _Float16 v16h;
typedef __attribute__((ext_vector_type(8)))  float    v8f;

union VF16 { v16h v; _Float16 h[16]; };
union VF8  { v8f  v; float    f[8];  };

// ---------------- WMMA A-fragment loader (gfx1250 wave32 layout) ----------------
// A 16x32 f16 (MxK): lane r (0-15) row M=r holds K {0..7,16..23}; lane r+16 same row, K {8..15,24..31}
// Per lane this is two contiguous 16B chunks -> two global_load_b128.
__device__ __forceinline__ v16h load_A_frag(const _Float16* __restrict__ A, int lda,
                                            int m0, int k0, int lane) {
  const int half = lane >> 4, r = lane & 15;
  const _Float16* row = A + (size_t)(m0 + r) * lda + k0;
  VF16 a;
#pragma unroll
  for (int j = 0; j < 8; ++j) a.h[j]     = row[half * 8 + j];        // K = half*8 .. +7
#pragma unroll
  for (int j = 0; j < 8; ++j) a.h[8 + j] = row[16 + half * 8 + j];   // K = 16+half*8 .. +7
  return a.v;
}

// ---------------- fused WMMA GEMM: out = act(A[f16] @ B[f16] + bias) ----------------
// A: MxK row-major (lda=K), B: KxN row-major (ldb=Ncols), out row-major MxN.
// 8 waves / block; each wave computes a 16x32 output tile (2 WMMA accumulators, A reused).
// B panel (32 cols) is pre-swizzled into LDS in *fragment order*:
//   Bs[((p*(K/32)+kb)*32 + lane)*16 + j]  => one aligned 32B v16h load per fragment (ds_load_b128 x2)
// Epilogue flags are template parameters -> straight-line stores, no per-element branches.
template <int KVAL, bool BIAS, bool RELU, bool WF, bool WH>
__global__ void wmma_gemm_kernel(const _Float16* __restrict__ A,
                                 const _Float16* __restrict__ B,
                                 const float* __restrict__ bias,
                                 float* __restrict__ outF,
                                 _Float16* __restrict__ outH,
                                 int M, int Ncols) {
  constexpr int KB = KVAL / 32;                       // k-blocks
  __shared__ __attribute__((aligned(32))) _Float16 Bs[2 * KB * 32 * 16];

  const int nbase = blockIdx.y * 32;
  // cooperative swizzled fill: consecutive threads hit consecutive columns (coalesced-ish)
  for (int idx = threadIdx.x; idx < 2 * KVAL * 16; idx += blockDim.x) {
    const int p    = idx / (KVAL * 16);
    const int rem  = idx - p * (KVAL * 16);
    const int kb   = rem >> 9;                        // / 512
    const int r    = rem & 511;
    const int j    = r >> 5;
    const int lane = r & 31;
    const int n    = p * 16 + (lane & 15);
    const int k    = kb * 32 + ((j >> 3) << 4) + ((lane >> 4) << 3) + (j & 7);
    Bs[((p * KB + kb) * 32 + lane) * 16 + j] = B[(size_t)k * Ncols + nbase + n];
  }
  __syncthreads();

  const int wave = threadIdx.x >> 5;
  const int lane = threadIdx.x & 31;
  const int mt   = blockIdx.x * (blockDim.x >> 5) + wave;
  const int m0   = mt * 16;
  if (m0 >= M) return;

  __builtin_prefetch(A + (size_t)(m0 + (lane & 15)) * KVAL, 0, 3);  // global_prefetch_b8, near

  v8f acc0 = {}, acc1 = {};
#pragma unroll
  for (int kb = 0; kb < KB; ++kb) {
    v16h a  = load_A_frag(A, KVAL, m0, kb * 32, lane);
    v16h b0 = *(const v16h*)(Bs + ((0 * KB + kb) * 32 + lane) * 16);
    v16h b1 = *(const v16h*)(Bs + ((1 * KB + kb) * 32 + lane) * 16);
    acc0 = __builtin_amdgcn_wmma_f32_16x16x32_f16(false, a, false, b0, (short)0, acc0, false, false);
    acc1 = __builtin_amdgcn_wmma_f32_16x16x32_f16(false, a, false, b1, (short)0, acc1, false, false);
  }

  // D layout: VGPR i -> row M=i (lanes<16) / M=8+i (lanes>=16), col N=lane%16
  const int half = lane >> 4, r = lane & 15;
  VF8 c0; c0.v = acc0;
  VF8 c1; c1.v = acc1;
#pragma unroll
  for (int p = 0; p < 2; ++p) {
    const int col  = nbase + p * 16 + r;
    const float bv = BIAS ? bias[col] : 0.0f;
    const VF8& c   = p ? c1 : c0;
#pragma unroll
    for (int i = 0; i < 8; ++i) {
      int row  = m0 + half * 8 + i;
      float v  = c.f[i];
      if (BIAS) v += bv;
      if (RELU) v = fmaxf(v, 0.f);
      size_t o = (size_t)row * Ncols + col;
      if (WF) outF[o] = v;
      if (WH) outH[o] = (_Float16)v;
    }
  }
}

// ---------------- LayerNorm(+ReLU)(+residual): one wave32 per node ----------------
__global__ void ln_relu_kernel(const float* __restrict__ in,
                               const float* __restrict__ g,
                               const float* __restrict__ b,
                               float* __restrict__ h,
                               _Float16* __restrict__ h16,
                               int residual) {
  const int wave = threadIdx.x >> 5, lane = threadIdx.x & 31;
  const int n = blockIdx.x * (blockDim.x >> 5) + wave;
  if (n >= NNODE) return;
  const size_t base = (size_t)n * HDIM;
  float v[4], s = 0.f;
#pragma unroll
  for (int j = 0; j < 4; ++j) { v[j] = in[base + j * 32 + lane]; s += v[j]; }
#pragma unroll
  for (int off = 16; off; off >>= 1) s += __shfl_xor(s, off, 32);
  const float mu = s * (1.0f / HDIM);
  float vs = 0.f;
#pragma unroll
  for (int j = 0; j < 4; ++j) { float d = v[j] - mu; vs += d * d; }
#pragma unroll
  for (int off = 16; off; off >>= 1) vs += __shfl_xor(vs, off, 32);
  const float rstd = rsqrtf(vs * (1.0f / HDIM) + LN_EPS_);
#pragma unroll
  for (int j = 0; j < 4; ++j) {
    int c = j * 32 + lane;
    float y = (v[j] - mu) * rstd * g[c] + b[c];
    y = fmaxf(y, 0.f);
    float hv = residual ? (h[base + c] + y) : y;
    h[base + c]   = hv;
    h16[base + c] = (_Float16)hv;
  }
}

// ---------------- attention scores: s = sum_c xp[n,h,c]*a[h,c], one wave per node ----------------
__global__ void attn_score_kernel(const float* __restrict__ xp,
                                  const float* __restrict__ asrc,
                                  const float* __restrict__ adst,
                                  float* __restrict__ ssrc,
                                  float* __restrict__ sdst) {
  const int wave = threadIdx.x >> 5, lane = threadIdx.x & 31;
  const int n = blockIdx.x * (blockDim.x >> 5) + wave;
  if (n >= NNODE) return;
  const size_t base = (size_t)n * HDIM;
#pragma unroll
  for (int hh = 0; hh < NHEAD; ++hh) {
    float x  = xp[base + hh * CDIM + lane];
    float vs = x * asrc[hh * CDIM + lane];
    float vd = x * adst[hh * CDIM + lane];
#pragma unroll
    for (int off = 16; off; off >>= 1) {
      vs += __shfl_xor(vs, off, 32);
      vd += __shfl_xor(vd, off, 32);
    }
    if (lane == 0) { ssrc[n * NHEAD + hh] = vs; sdst[n * NHEAD + hh] = vd; }
  }
}

// ---------------- edge helpers ----------------
__device__ __forceinline__ void get_edge(const int* __restrict__ ei, int e, int& s, int& d) {
  if (e < NEDGE) { s = ei[e]; d = ei[NEDGE + e]; } else { s = e - NEDGE; d = s; }
}
// monotonic float <-> uint encoding for atomicMax on signed floats
__device__ __forceinline__ unsigned f32key(float f) {
  unsigned b = __float_as_uint(f);
  return (b & 0x80000000u) ? ~b : (b | 0x80000000u);
}
__device__ __forceinline__ float keyf32(unsigned k) {
  unsigned b = (k & 0x80000000u) ? (k & 0x7FFFFFFFu) : ~k;
  return __uint_as_float(b);
}
#define NEGINF_KEY 0x007FFFFFu   // f32key(-inf)
__device__ __forceinline__ float lrelu(float a) { return a > 0.f ? a : NEG_SLOPE_ * a; }

__global__ void edge_amax_kernel(const int* __restrict__ ei,
                                 const float* __restrict__ ssrc,
                                 const float* __restrict__ sdst,
                                 unsigned* __restrict__ amax) {
  int e = blockIdx.x * blockDim.x + threadIdx.x;
  if (e >= ETOT) return;
  int s, d; get_edge(ei, e, s, d);
#pragma unroll
  for (int hh = 0; hh < NHEAD; ++hh) {
    float a = lrelu(ssrc[s * NHEAD + hh] + sdst[d * NHEAD + hh]);
    atomicMax(&amax[d * NHEAD + hh], f32key(a));
  }
}

__global__ void edge_ex_kernel(const int* __restrict__ ei,
                               const float* __restrict__ ssrc,
                               const float* __restrict__ sdst,
                               const unsigned* __restrict__ amax,
                               float* __restrict__ ex,
                               float* __restrict__ den) {
  int e = blockIdx.x * blockDim.x + threadIdx.x;
  if (e >= ETOT) return;
  int s, d; get_edge(ei, e, s, d);
#pragma unroll
  for (int hh = 0; hh < NHEAD; ++hh) {
    float a  = lrelu(ssrc[s * NHEAD + hh] + sdst[d * NHEAD + hh]);
    float ev = expf(a - keyf32(amax[d * NHEAD + hh]));
    ex[(size_t)e * NHEAD + hh] = ev;
    atomicAdd(&den[d * NHEAD + hh], ev);
  }
}

// 32 lanes per edge: lane covers dim hh*32+lane for each head hh
__global__ void edge_agg_kernel(const int* __restrict__ ei,
                                const float* __restrict__ xp,
                                const float* __restrict__ ex,
                                const float* __restrict__ den,
                                float* __restrict__ agg) {
  int gid = blockIdx.x * blockDim.x + threadIdx.x;
  int e = gid >> 5, lane = gid & 31;
  if (e >= ETOT) return;
  int s, d; get_edge(ei, e, s, d);
#pragma unroll
  for (int hh = 0; hh < NHEAD; ++hh) {
    float attn = ex[(size_t)e * NHEAD + hh] / den[d * NHEAD + hh];
    atomicAdd(&agg[(size_t)d * HDIM + hh * CDIM + lane],
              xp[(size_t)s * HDIM + hh * CDIM + lane] * attn);
  }
}

// ---------------- small utility kernels ----------------
__global__ void fill_u32_kernel(unsigned* __restrict__ p, unsigned v, int n) {
  int i = blockIdx.x * blockDim.x + threadIdx.x;
  if (i < n) p[i] = v;
}
__global__ void cvt_pad_kernel(const float* __restrict__ s, _Float16* __restrict__ d,
                               int ntot, int nvalid) {
  int i = blockIdx.x * blockDim.x + threadIdx.x;
  if (i < ntot) d[i] = (i < nvalid) ? (_Float16)s[i] : (_Float16)0.f;
}
__global__ void build_xcat_kernel(const float* __restrict__ x, const float* __restrict__ act,
                                  _Float16* __restrict__ xc) {
  int i = blockIdx.x * blockDim.x + threadIdx.x;
  if (i >= NNODE * KENC) return;
  int n = i >> 6, j = i & (KENC - 1);
  float v = 0.f;
  if (j < FEAT_)             v = x[n * FEAT_ + j];
  else if (j < FEAT_ + ACT_) v = act[n * ACT_ + (j - FEAT_)];
  xc[i] = (_Float16)v;
}
__global__ void agg_init_kernel(float* __restrict__ agg, const float* __restrict__ bias) {
  int i = blockIdx.x * blockDim.x + threadIdx.x;
  if (i < NNODE * HDIM) agg[i] = bias[i & (HDIM - 1)];
}
__global__ void mean_pool_kernel(const float* __restrict__ h, float* __restrict__ gemb) {
  int i = blockIdx.x * blockDim.x + threadIdx.x;
  if (i < NNODE * HDIM) atomicAdd(&gemb[i & (HDIM - 1)], h[i] * (1.0f / NNODE));
}
__global__ void add_x_kernel(const float* __restrict__ x, const float* __restrict__ delta,
                             float* __restrict__ nx) {
  int i = blockIdx.x * blockDim.x + threadIdx.x;
  if (i < NNODE * FEAT_) nx[i] = x[i] + delta[i];
}
__global__ void heads_kernel(const float* __restrict__ gemb,
                             const float* __restrict__ Wr1, const float* __restrict__ br1,
                             const float* __restrict__ Wr2, const float* __restrict__ br2,
                             const float* __restrict__ Wc1, const float* __restrict__ bc1,
                             const float* __restrict__ Wc2, const float* __restrict__ bc2,
                             float* __restrict__ out2) {
  __shared__ float r1[64], c1[64];
  int j = threadIdx.x;                       // 64 threads
  float sr = br1[j], sc = bc1[j];
  for (int k = 0; k < HDIM; ++k) {
    float gv = gemb[k];
    sr += gv * Wr1[k * 64 + j];
    sc += gv * Wc1[k * 64 + j];
  }
  r1[j] = fmaxf(sr, 0.f);
  c1[j] = fmaxf(sc, 0.f);
  __syncthreads();
  if (j == 0) {
    float r = br2[0];
    for (int k = 0; k < 64; ++k) r += r1[k] * Wr2[k];
    out2[0] = r;
  } else if (j == 1) {
    float c = bc2[0];
    for (int k = 0; k < 64; ++k) c += c1[k] * Wc2[k];
    out2[1] = 1.0f / (1.0f + expf(-c));
  }
}

// ---------------- launch ----------------
extern "C" void kernel_launch(void* const* d_in, const int* in_sizes, int n_in,
                              void* d_out, int out_size, void* d_ws, size_t ws_size,
                              hipStream_t stream) {
  (void)in_sizes; (void)n_in; (void)out_size; (void)ws_size;
  const float* x      = (const float*)d_in[0];
  const float* action = (const float*)d_in[1];
  const int*   ei     = (const int*)d_in[2];
  const float* W0     = (const float*)d_in[3];
  const float* b0     = (const float*)d_in[4];
  const float* ln0_g  = (const float*)d_in[5];
  const float* ln0_b  = (const float*)d_in[6];
  const float* Wg     = (const float*)d_in[7];
  const float* att_s  = (const float*)d_in[8];
  const float* att_d  = (const float*)d_in[9];
  const float* bg     = (const float*)d_in[10];
  const float* lng_g  = (const float*)d_in[11];
  const float* lng_b  = (const float*)d_in[12];
  const float* Wd1    = (const float*)d_in[13];
  const float* bd1    = (const float*)d_in[14];
  const float* Wd2    = (const float*)d_in[15];
  const float* bd2    = (const float*)d_in[16];
  const float* Wr1    = (const float*)d_in[17];
  const float* br1    = (const float*)d_in[18];
  const float* Wr2    = (const float*)d_in[19];
  const float* br2    = (const float*)d_in[20];
  const float* Wc1    = (const float*)d_in[21];
  const float* bc1    = (const float*)d_in[22];
  const float* Wc2    = (const float*)d_in[23];
  const float* bc2    = (const float*)d_in[24];
  float* dout = (float*)d_out;

  // workspace carve-out (256B aligned)
  char* base = (char*)d_ws;
  size_t off = 0;
  auto alloc = [&](size_t bytes) -> char* {
    char* p = base + off;
    off += (bytes + 255) & ~(size_t)255;
    return p;
  };
  float*    h      = (float*)   alloc((size_t)NNODE * HDIM * 4);
  _Float16* h16    = (_Float16*)alloc((size_t)NNODE * HDIM * 2);
  float*    xpb    = (float*)   alloc((size_t)NNODE * HDIM * 4);
  float*    agg    = (float*)   alloc((size_t)NNODE * HDIM * 4);
  _Float16* xcat16 = (_Float16*)alloc((size_t)NNODE * KENC * 2);
  _Float16* hid16  = (_Float16*)alloc((size_t)NNODE * HDIM * 2);
  float*    ssrc   = (float*)   alloc((size_t)NNODE * NHEAD * 4);
  float*    sdst   = (float*)   alloc((size_t)NNODE * NHEAD * 4);
  float*    den    = (float*)   alloc((size_t)NNODE * NHEAD * 4);
  unsigned* amax   = (unsigned*)alloc((size_t)NNODE * NHEAD * 4);
  float*    ex     = (float*)   alloc((size_t)ETOT  * NHEAD * 4);
  _Float16* W0_16  = (_Float16*)alloc((size_t)KENC * HDIM * 2);
  _Float16* Wg16   = (_Float16*)alloc((size_t)3 * HDIM * HDIM * 2);
  _Float16* Wd1_16 = (_Float16*)alloc((size_t)HDIM * HDIM * 2);
  _Float16* Wd2_16 = (_Float16*)alloc((size_t)HDIM * FEAT_ * 2);
  float*    gemb   = (float*)   alloc(128 * 4);

  auto cdiv = [](long a, long b) { return (int)((a + b - 1) / b); };
  const int T = 256;

  // weight conversion to f16 (W0 zero-padded to 64 rows)
  cvt_pad_kernel<<<cdiv(KENC * HDIM, T), T, 0, stream>>>(W0, W0_16, KENC * HDIM, 40 * HDIM);
  cvt_pad_kernel<<<cdiv(3 * HDIM * HDIM, T), T, 0, stream>>>(Wg, Wg16, 3 * HDIM * HDIM, 3 * HDIM * HDIM);
  cvt_pad_kernel<<<cdiv(HDIM * HDIM, T), T, 0, stream>>>(Wd1, Wd1_16, HDIM * HDIM, HDIM * HDIM);
  cvt_pad_kernel<<<cdiv(HDIM * FEAT_, T), T, 0, stream>>>(Wd2, Wd2_16, HDIM * FEAT_, HDIM * FEAT_);
  build_xcat_kernel<<<cdiv((long)NNODE * KENC, T), T, 0, stream>>>(x, action, xcat16);

  const int MT = NNODE / 16;                 // 3125 row tiles
  dim3 gemmGridH(cdiv(MT, 8), HDIM / 32);    // 128 output cols -> 4 panels of 32
  dim3 gemmGridF(cdiv(MT, 8), FEAT_ / 32);   // 32 output cols  -> 1 panel

  // encoder: xp = xcat @ W0 + b0 ; h = relu(LN(xp))
  wmma_gemm_kernel<KENC, true, false, true, false>
      <<<gemmGridH, T, 0, stream>>>(xcat16, W0_16, b0, xpb, nullptr, NNODE, HDIM);
  ln_relu_kernel<<<cdiv(NNODE, 4), 128, 0, stream>>>(xpb, ln0_g, ln0_b, h, h16, 0);

  // GAT layers
  for (int l = 0; l < 3; ++l) {
    wmma_gemm_kernel<HDIM, false, false, true, false>
        <<<gemmGridH, T, 0, stream>>>(h16, Wg16 + (size_t)l * HDIM * HDIM,
                                      nullptr, xpb, nullptr, NNODE, HDIM);
    attn_score_kernel<<<cdiv(NNODE, 4), 128, 0, stream>>>(
        xpb, att_s + (size_t)l * NHEAD * CDIM, att_d + (size_t)l * NHEAD * CDIM, ssrc, sdst);
    fill_u32_kernel<<<cdiv(NNODE * NHEAD, T), T, 0, stream>>>(amax, NEGINF_KEY, NNODE * NHEAD);
    fill_u32_kernel<<<cdiv(NNODE * NHEAD, T), T, 0, stream>>>((unsigned*)den, 0u, NNODE * NHEAD);
    agg_init_kernel<<<cdiv((long)NNODE * HDIM, T), T, 0, stream>>>(agg, bg + (size_t)l * HDIM);
    edge_amax_kernel<<<cdiv(ETOT, T), T, 0, stream>>>(ei, ssrc, sdst, amax);
    edge_ex_kernel<<<cdiv(ETOT, T), T, 0, stream>>>(ei, ssrc, sdst, amax, ex, den);
    edge_agg_kernel<<<cdiv((long)ETOT * 32, T), T, 0, stream>>>(ei, xpb, ex, den, agg);
    ln_relu_kernel<<<cdiv(NNODE, 4), 128, 0, stream>>>(
        agg, lng_g + (size_t)l * HDIM, lng_b + (size_t)l * HDIM, h, h16, 1);
  }

  // dynamics head: hid = relu(h @ Wd1 + bd1); delta = hid @ Wd2 + bd2; next_x = x + delta
  wmma_gemm_kernel<HDIM, true, true, false, true>
      <<<gemmGridH, T, 0, stream>>>(h16, Wd1_16, bd1, nullptr, hid16, NNODE, HDIM);
  wmma_gemm_kernel<HDIM, true, false, true, false>
      <<<gemmGridF, T, 0, stream>>>(hid16, Wd2_16, bd2, dout, nullptr, NNODE, FEAT_);
  add_x_kernel<<<cdiv((long)NNODE * FEAT_, T), T, 0, stream>>>(x, dout, dout + (size_t)NNODE * FEAT_);

  // graph mean pool + reward / constraint heads
  fill_u32_kernel<<<1, 128, 0, stream>>>((unsigned*)gemb, 0u, 128);
  mean_pool_kernel<<<cdiv((long)NNODE * HDIM, T), T, 0, stream>>>(h, gemb);
  heads_kernel<<<1, 64, 0, stream>>>(gemb, Wr1, br1, Wr2, br2, Wc1, bc1, Wc2, bc2,
                                     dout + (size_t)2 * NNODE * FEAT_);
}
